// RaffelAttention_69758858821738
// MI455X (gfx1250) — compile-verified, matrix operand order checked by reading, not run on hardware
//
#include <hip/hip_runtime.h>

// ---------------------------------------------------------------------------
// Disentangled attention for MI455X (gfx1250, wave32, WMMA).
// All matmuls use v_wmma_f32_16x16x32_bf16 (f32 accumulate, bf16 A/B staged
// through LDS). The 512MB attention-probability output makes this workload
// HBM-bound (~2GB attn traffic => ~90us @ 23.3 TB/s), so bf16 WMMA is more
// than enough compute while minimizing operand bytes.
//
// GEMM K-loops are software-pipelined: next tile's global loads are issued
// into registers while WMMAs run on the current double-buffered LDS tile;
// one workgroup barrier per K-step.
// ---------------------------------------------------------------------------

#define B_   2
#define L_   2048
#define D_   1024
#define H_   16
#define HD_  64

typedef __attribute__((ext_vector_type(16))) __bf16        v16bf;
typedef __attribute__((ext_vector_type(8)))  float         v8f;
typedef __attribute__((ext_vector_type(4)))  float         f32x4;
typedef __attribute__((ext_vector_type(4)))  unsigned int  u32x4;
typedef __attribute__((ext_vector_type(8)))  unsigned int  u32x8;

__device__ __forceinline__ unsigned short f2bf(float x) {
    unsigned int u = __builtin_bit_cast(unsigned int, x);
    u += 0x7FFFu + ((u >> 16) & 1u);         // round-to-nearest-even
    return (unsigned short)(u >> 16);
}

// Load a 16x32 bf16 A/B fragment from LDS (row-major [row][ldk] ushorts).
// Lane holds row/col (lane&15); K chunks: lanes<16 -> {0..7,16..23},
// lanes>=16 -> {8..15,24..31} per the CDNA5 ISA 16-bit A-matrix layout.
__device__ __forceinline__ v16bf load_frag(const unsigned short* base, int row,
                                           int ldk, int kbase, int lane) {
    const unsigned short* p = base + row * ldk + kbase + ((lane & 16) ? 8 : 0);
    u32x4 a = *(const u32x4*)(p);
    u32x4 b = *(const u32x4*)(p + 16);
    u32x8 r = {a[0], a[1], a[2], a[3], b[0], b[1], b[2], b[3]};
    return __builtin_bit_cast(v16bf, r);
}

__device__ __forceinline__ v8f wmma_bf16(v16bf a, v16bf b, v8f c) {
    return __builtin_amdgcn_wmma_f32_16x16x32_bf16(
        /*neg_a=*/false, a, /*neg_b=*/false, b,
        /*c_mod=*/(short)0, c, /*reuse_a=*/false, /*reuse_b=*/false);
}

// ---------------------------------------------------------------------------
// Kernel 1: Y(bf16)[4096,1024] = X(f32)[4096,1024] @ W(f32)[1024,1024] + bias
// Block tile 128x64, 8 waves; double-buffered LDS, pipelined global loads.
// ---------------------------------------------------------------------------
__global__ __launch_bounds__(256) void proj_kernel(const float* __restrict__ X,
                                                   const float* __restrict__ W,
                                                   const float* __restrict__ bias,
                                                   unsigned short* __restrict__ Y) {
    constexpr int TM = 128, TN = 64, TK = 32;
    __shared__ unsigned short As[2][TM * TK];   // [m][k] bf16
    __shared__ unsigned short Bs[2][TN * TK];   // W^T tile: [n][k] bf16
    const int row0 = blockIdx.x * TM;
    const int col0 = blockIdx.y * TN;
    const int tid = threadIdx.x, lane = tid & 31, wave = tid >> 5;

    f32x4 ra[4];   // A staging regs: 128*32/4 = 1024 f32x4 / 256 thr = 4
    f32x4 rb[2];   // B staging regs: 32*64/4  =  512 f32x4 / 256 thr = 2

    auto issue = [&](int kb) {
        for (int j = 0; j < 4; ++j) {
            int i = tid + j * 256, r = i >> 3, c = (i & 7) * 4;
            ra[j] = *(const f32x4*)(X + (size_t)(row0 + r) * D_ + kb + c);
        }
        for (int j = 0; j < 2; ++j) {
            int i = tid + j * 256, k = i >> 4, n = (i & 15) * 4;
            rb[j] = *(const f32x4*)(W + (size_t)(kb + k) * D_ + col0 + n);
        }
    };
    auto store = [&](int buf) {
        for (int j = 0; j < 4; ++j) {
            int i = tid + j * 256, r = i >> 3, c = (i & 7) * 4;
            unsigned short* d = As[buf] + r * TK + c;
            d[0] = f2bf(ra[j][0]); d[1] = f2bf(ra[j][1]);
            d[2] = f2bf(ra[j][2]); d[3] = f2bf(ra[j][3]);
        }
        for (int j = 0; j < 2; ++j) {
            int i = tid + j * 256, k = i >> 4, n = (i & 15) * 4;
            Bs[buf][(n + 0) * TK + k] = f2bf(rb[j][0]);
            Bs[buf][(n + 1) * TK + k] = f2bf(rb[j][1]);
            Bs[buf][(n + 2) * TK + k] = f2bf(rb[j][2]);
            Bs[buf][(n + 3) * TK + k] = f2bf(rb[j][3]);
        }
    };

    v8f acc[4] = {};
    const int m = lane & 15;
    issue(0);
    for (int kb = 0, it = 0; kb < D_; kb += TK, ++it) {
        const int cur = it & 1;
        store(cur);
        __syncthreads();
        if (kb + TK < D_) issue(kb + TK);          // overlap with WMMAs below
        v16bf a = load_frag(As[cur], wave * 16 + m, TK, 0, lane);
        for (int t = 0; t < 4; ++t) {
            v16bf b = load_frag(Bs[cur], t * 16 + m, TK, 0, lane);
            acc[t] = wmma_bf16(a, b, acc[t]);
        }
    }
    // writeback: lane holds n=lane&15, rows (lane/16)*8 + r
    const int n = lane & 15, mb = (lane >> 4) * 8;
    for (int t = 0; t < 4; ++t) {
        int col = col0 + t * 16 + n;
        float bv = bias[col];
        for (int r = 0; r < 8; ++r) {
            int row = row0 + wave * 16 + mb + r;
            Y[(size_t)row * D_ + col] = f2bf(acc[t][r] + bv);
        }
    }
}

// ---------------------------------------------------------------------------
// Kernel 2: raw scores S[b,h,q,k] = cq.ck^T + pq.pk^T + rel_bias[h]  (f32)
// Block: 64q x 64k tile per (b,h). 8 waves x 2 subtiles = 16 16x16 tiles.
// ---------------------------------------------------------------------------
__global__ __launch_bounds__(256) void scores_kernel(const unsigned short* __restrict__ cq,
                                                     const unsigned short* __restrict__ ck,
                                                     const unsigned short* __restrict__ pq,
                                                     const unsigned short* __restrict__ pk,
                                                     const float* __restrict__ rel_bias,
                                                     float* __restrict__ attn) {
    __shared__ unsigned short sCQ[64 * HD_], sPQ[64 * HD_], sCK[64 * HD_], sPK[64 * HD_];
    const int bh = blockIdx.z, b = bh >> 4, h = bh & 15;
    const int q0 = blockIdx.y * 64, k0 = blockIdx.x * 64;
    const size_t baseQ = ((size_t)b * L_ + q0) * D_ + h * HD_;
    const size_t baseK = ((size_t)b * L_ + k0) * D_ + h * HD_;
    const int tid = threadIdx.x;

    // stage 4 tiles of 64 rows x 64 bf16
    for (int i = tid; i < 64 * HD_ / 8; i += 256) {
        int r = i >> 3, c = (i & 7) * 8;
        *(u32x4*)(sCQ + r * HD_ + c) = *(const u32x4*)(cq + baseQ + (size_t)r * D_ + c);
        *(u32x4*)(sPQ + r * HD_ + c) = *(const u32x4*)(pq + baseQ + (size_t)r * D_ + c);
        *(u32x4*)(sCK + r * HD_ + c) = *(const u32x4*)(ck + baseK + (size_t)r * D_ + c);
        *(u32x4*)(sPK + r * HD_ + c) = *(const u32x4*)(pk + baseK + (size_t)r * D_ + c);
    }
    __syncthreads();

    const int lane = tid & 31, wave = tid >> 5;
    const int qsub = wave >> 1, m = lane & 15;
    v8f acc[2] = {};
    for (int kk = 0; kk < HD_; kk += 32) {
        v16bf aC = load_frag(sCQ, qsub * 16 + m, HD_, kk, lane);
        v16bf aP = load_frag(sPQ, qsub * 16 + m, HD_, kk, lane);
        for (int t = 0; t < 2; ++t) {
            int ksub = (wave & 1) * 2 + t;
            v16bf bC = load_frag(sCK, ksub * 16 + m, HD_, kk, lane);
            acc[t] = wmma_bf16(aC, bC, acc[t]);
            v16bf bP = load_frag(sPK, ksub * 16 + m, HD_, kk, lane);
            acc[t] = wmma_bf16(aP, bP, acc[t]);
        }
    }
    const float rb = rel_bias[h];
    const int n = lane & 15, mb = (lane >> 4) * 8;
    const size_t outBase = ((size_t)bh * L_ + q0) * L_ + k0;
    for (int t = 0; t < 2; ++t) {
        int ksub = (wave & 1) * 2 + t;
        for (int r = 0; r < 8; ++r) {
            int q = qsub * 16 + mb + r;
            attn[outBase + (size_t)q * L_ + ksub * 16 + n] = acc[t][r] + rb;
        }
    }
}

// ---------------------------------------------------------------------------
// Kernel 3: in-place row softmax of attn (scale 1/sqrt(64)); 1 block / row.
// ---------------------------------------------------------------------------
__global__ __launch_bounds__(256) void softmax_kernel(float* __restrict__ attn) {
    float* p = attn + (size_t)blockIdx.x * L_;
    const int tid = threadIdx.x;
    __shared__ float red[256];

    float x[8];
    float mx = -3.402823466e38f;
    for (int i = 0; i < 8; ++i) {
        x[i] = p[tid + i * 256] * 0.125f;
        mx = fmaxf(mx, x[i]);
    }
    red[tid] = mx; __syncthreads();
    for (int s = 128; s > 0; s >>= 1) {
        if (tid < s) red[tid] = fmaxf(red[tid], red[tid + s]);
        __syncthreads();
    }
    mx = red[0]; __syncthreads();

    float sum = 0.f;
    for (int i = 0; i < 8; ++i) { x[i] = __expf(x[i] - mx); sum += x[i]; }
    red[tid] = sum; __syncthreads();
    for (int s = 128; s > 0; s >>= 1) {
        if (tid < s) red[tid] += red[tid + s];
        __syncthreads();
    }
    const float inv = 1.0f / red[0];
    for (int i = 0; i < 8; ++i) p[tid + i * 256] = x[i] * inv;
}

// ---------------------------------------------------------------------------
// Kernel 4: ctx[b,q,h*64+d] (bf16) = attn[b,h,q,:] @ cv[b,:,h*64+d], cv = ck.
// Block: 64 q-rows x full HD=64 per (b,h); pipelined k-loop, steps of 32.
// ---------------------------------------------------------------------------
__global__ __launch_bounds__(256) void ctx_kernel(const float* __restrict__ attn,
                                                  const unsigned short* __restrict__ ck,
                                                  unsigned short* __restrict__ ctx) {
    __shared__ unsigned short sA[2][64 * 32];   // attn tile bf16 [q][k]
    __shared__ unsigned short sV[2][64 * 32];   // cv^T tile     [d][k]
    const int bh = blockIdx.y, b = bh >> 4, h = bh & 15;
    const int q0 = blockIdx.x * 64;
    const int tid = threadIdx.x, lane = tid & 31, wave = tid >> 5;
    const int qsub = wave >> 1, m = lane & 15;
    const size_t attnBase = ((size_t)bh * L_ + q0) * L_;
    const size_t vBase = (size_t)b * L_ * D_ + h * HD_;

    f32x4 rA[2];            // 64*32/4 = 512 f32x4 / 256 thr = 2
    unsigned short rV[8];   // 32*64 ushorts / 256 thr = 8

    auto issue = [&](int k0) {
        for (int j = 0; j < 2; ++j) {
            int i = tid + j * 256, r = i >> 3, c = (i & 7) * 4;
            rA[j] = *(const f32x4*)(attn + attnBase + (size_t)r * L_ + k0 + c);
        }
        for (int j = 0; j < 8; ++j) {
            int i = tid + j * 256, k = i >> 6, d = i & 63;
            rV[j] = ck[vBase + (size_t)(k0 + k) * D_ + d];
        }
    };
    auto store = [&](int buf) {
        for (int j = 0; j < 2; ++j) {
            int i = tid + j * 256, r = i >> 3, c = (i & 7) * 4;
            unsigned short* d = sA[buf] + r * 32 + c;
            d[0] = f2bf(rA[j][0]); d[1] = f2bf(rA[j][1]);
            d[2] = f2bf(rA[j][2]); d[3] = f2bf(rA[j][3]);
        }
        for (int j = 0; j < 8; ++j) {
            int i = tid + j * 256, k = i >> 6, d = i & 63;
            sV[buf][d * 32 + k] = rV[j];
        }
    };

    v8f acc[2] = {};
    issue(0);
    for (int k0 = 0, it = 0; k0 < L_; k0 += 32, ++it) {
        const int cur = it & 1;
        store(cur);
        __syncthreads();
        if (k0 + 32 < L_) issue(k0 + 32);          // overlap with WMMAs below
        v16bf a = load_frag(sA[cur], qsub * 16 + m, 32, 0, lane);
        for (int t = 0; t < 2; ++t) {
            int dsub = (wave & 1) * 2 + t;
            v16bf bv = load_frag(sV[cur], dsub * 16 + m, 32, 0, lane);
            acc[t] = wmma_bf16(a, bv, acc[t]);
        }
    }
    const int n = lane & 15, mb = (lane >> 4) * 8;
    for (int t = 0; t < 2; ++t) {
        int col = h * HD_ + ((wave & 1) * 2 + t) * 16 + n;
        for (int r = 0; r < 8; ++r) {
            int q = q0 + qsub * 16 + mb + r;
            ctx[((size_t)b * L_ + q) * D_ + col] = f2bf(acc[t][r]);
        }
    }
}

// ---------------------------------------------------------------------------
// Kernel 5: out(f32)[4096,1024] = ctx(bf16) @ Wo(f32) + bo ; pipelined.
// ---------------------------------------------------------------------------
__global__ __launch_bounds__(256) void out_kernel(const unsigned short* __restrict__ Xb,
                                                  const float* __restrict__ W,
                                                  const float* __restrict__ bias,
                                                  float* __restrict__ Y) {
    constexpr int TM = 128, TN = 64, TK = 32;
    __shared__ unsigned short As[2][TM * TK];
    __shared__ unsigned short Bs[2][TN * TK];
    const int row0 = blockIdx.x * TM;
    const int col0 = blockIdx.y * TN;
    const int tid = threadIdx.x, lane = tid & 31, wave = tid >> 5;

    u32x4 ra[2];   // A (bf16): 128*32 ushorts = 512 u32x4 / 256 thr = 2
    f32x4 rb[2];   // B (f32) : 32*64/4 = 512 f32x4 / 256 thr = 2

    auto issue = [&](int kb) {
        for (int j = 0; j < 2; ++j) {
            int i = tid + j * 256, r = i >> 2, c = (i & 3) * 8;
            ra[j] = *(const u32x4*)(Xb + (size_t)(row0 + r) * D_ + kb + c);
        }
        for (int j = 0; j < 2; ++j) {
            int i = tid + j * 256, k = i >> 4, n = (i & 15) * 4;
            rb[j] = *(const f32x4*)(W + (size_t)(kb + k) * D_ + col0 + n);
        }
    };
    auto store = [&](int buf) {
        for (int j = 0; j < 2; ++j) {
            int i = tid + j * 256, r = i >> 2, c = (i & 3) * 8;
            *(u32x4*)(As[buf] + r * TK + c) = ra[j];
        }
        for (int j = 0; j < 2; ++j) {
            int i = tid + j * 256, k = i >> 4, n = (i & 15) * 4;
            Bs[buf][(n + 0) * TK + k] = f2bf(rb[j][0]);
            Bs[buf][(n + 1) * TK + k] = f2bf(rb[j][1]);
            Bs[buf][(n + 2) * TK + k] = f2bf(rb[j][2]);
            Bs[buf][(n + 3) * TK + k] = f2bf(rb[j][3]);
        }
    };

    v8f acc[4] = {};
    const int m = lane & 15;
    issue(0);
    for (int kb = 0, it = 0; kb < D_; kb += TK, ++it) {
        const int cur = it & 1;
        store(cur);
        __syncthreads();
        if (kb + TK < D_) issue(kb + TK);
        v16bf a = load_frag(As[cur], wave * 16 + m, TK, 0, lane);
        for (int t = 0; t < 4; ++t) {
            v16bf b = load_frag(Bs[cur], t * 16 + m, TK, 0, lane);
            acc[t] = wmma_bf16(a, b, acc[t]);
        }
    }
    const int n = lane & 15, mb = (lane >> 4) * 8;
    for (int t = 0; t < 4; ++t) {
        int col = col0 + t * 16 + n;
        float bv = bias[col];
        for (int r = 0; r < 8; ++r) {
            int row = row0 + wave * 16 + mb + r;
            Y[(size_t)row * D_ + col] = acc[t][r] + bv;
        }
    }
}

// ---------------------------------------------------------------------------
extern "C" void kernel_launch(void* const* d_in, const int* in_sizes, int n_in,
                              void* d_out, int out_size, void* d_ws, size_t ws_size,
                              hipStream_t stream) {
    (void)in_sizes; (void)n_in; (void)out_size; (void)ws_size;

    const float* hs  = (const float*)d_in[0];
    const float* pe  = (const float*)d_in[1];
    const float* Wcq = (const float*)d_in[2];
    const float* bcq = (const float*)d_in[3];
    const float* Wck = (const float*)d_in[4];
    const float* bck = (const float*)d_in[5];
    const float* Wpq = (const float*)d_in[6];
    const float* bpq = (const float*)d_in[7];
    const float* Wpk = (const float*)d_in[8];
    const float* bpk = (const float*)d_in[9];
    const float* rb  = (const float*)d_in[10];
    const float* Wo  = (const float*)d_in[11];
    const float* bo  = (const float*)d_in[12];

    float* out  = (float*)d_out;                       // [B, L, D]
    float* attn = out + (size_t)B_ * L_ * D_;          // [B, H, L, L]

    const size_t S = (size_t)B_ * L_ * D_;             // elements per bf16 buffer
    unsigned short* ws = (unsigned short*)d_ws;
    unsigned short* cq  = ws + 0 * S;
    unsigned short* ck  = ws + 1 * S;
    unsigned short* pq  = ws + 2 * S;
    unsigned short* pk  = ws + 3 * S;
    unsigned short* ctx = ws + 4 * S;

    const dim3 gp(B_ * L_ / 128, D_ / 64);             // 32 x 16
    proj_kernel<<<gp, 256, 0, stream>>>(hs, Wcq, bcq, cq);
    proj_kernel<<<gp, 256, 0, stream>>>(hs, Wck, bck, ck);
    proj_kernel<<<gp, 256, 0, stream>>>(pe, Wpq, bpq, pq);
    proj_kernel<<<gp, 256, 0, stream>>>(pe, Wpk, bpk, pk);

    scores_kernel<<<dim3(L_ / 64, L_ / 64, B_ * H_), 256, 0, stream>>>(
        cq, ck, pq, pk, rb, attn);

    softmax_kernel<<<(unsigned)(B_ * H_ * L_), 256, 0, stream>>>(attn);

    ctx_kernel<<<dim3(L_ / 64, B_ * H_), 256, 0, stream>>>(attn, ck, ctx);

    out_kernel<<<gp, 256, 0, stream>>>(ctx, Wo, bo, out);
}
